// SelfAttention_1219770712742
// MI455X (gfx1250) — compile-verified, matrix operand order checked by reading, not run on hardware
//
#include <hip/hip_runtime.h>
#include <hip/hip_bf16.h>

typedef _Float16 f16;
typedef __attribute__((ext_vector_type(16))) _Float16 v16h;
typedef __attribute__((ext_vector_type(8)))  _Float16 v8h;
typedef __attribute__((ext_vector_type(8)))  float    v8f;

#define BB   4
#define SS   2048
#define FF   2048
#define HH   16
#define GG   4
#define DH   128
#define MM   (BB * SS)          // 8192 token rows

// ---------------- workspace layout (bytes) ----------------
// attn16 aliases x16 (x16 is dead after the QKV GEMMs).
static const size_t OFF_X16  = 0;              // 8192*2048*2  = 33,554,432 (also attn16)
static const size_t OFF_WQT  = 33554432;       // 2048*2048*2  =  8,388,608
static const size_t OFF_WKT  = 41943040;       //  512*2048*2  =  2,097,152
static const size_t OFF_WVT  = 44040192;       //  512*2048*2  =  2,097,152
static const size_t OFF_WOT  = 46137344;       // 2048*2048*2  =  8,388,608
static const size_t OFF_QRAW = 54525952;       // 8192*2048*2  = 33,554,432
static const size_t OFF_KRAW = 88080384;       // 8192* 512*2  =  8,388,608
static const size_t OFF_VRAW = 96468992;       // 8192* 512*2  =  8,388,608
static const size_t OFF_Q16  = 104857600;      // 33,554,432
static const size_t OFF_K16  = 138412032;      //  8,388,608
static const size_t OFF_VT16 = 146800640;      //  8,388,608  -> end 155,189,248

__device__ __forceinline__ v16h pack16(v8h lo, v8h hi) {
    v16h r;
#pragma unroll
    for (int i = 0; i < 8; ++i) { r[i] = lo[i]; r[i + 8] = hi[i]; }
    return r;
}

// ---------------- prep kernels ----------------
__global__ void cvt_f32_to_f16(const float* __restrict__ src, f16* __restrict__ dst, size_t n) {
    size_t i = (size_t)blockIdx.x * blockDim.x + threadIdx.x;
    if (i < n) dst[i] = (f16)src[i];
}

// src: f32 [R][C]  ->  dst: f16 [C][R]
__global__ void wtrans_f32_to_f16(const float* __restrict__ src, f16* __restrict__ dst,
                                  int R, int C) {
    size_t i = (size_t)blockIdx.x * blockDim.x + threadIdx.x;
    size_t n = (size_t)R * C;
    if (i >= n) return;
    int c = (int)(i % C);
    int r = (int)(i / C);
    dst[(size_t)c * R + r] = (f16)src[i];
}

// ---------------- WMMA GEMM: C[M][N] = A[M][K](f16) * BT[N][K](f16) + bias[N] ----------------
// block = 128 threads (4 waves); each wave computes one 16x16 tile.
template <typename CT>
__global__ void __launch_bounds__(128, 1)
gemm_wmma(const f16* __restrict__ A, const f16* __restrict__ BT,
          const float* __restrict__ bias, CT* __restrict__ C,
          int M, int N, int K) {
    const int lane = threadIdx.x & 31;
    const int wave = threadIdx.x >> 5;
    const int n0   = blockIdx.x * 16;
    const int m0   = (blockIdx.y * 4 + wave) * 16;
    const int col  = lane & 15;
    const int kb8  = (lane >> 4) * 8;

    const f16* ap = A  + (size_t)(m0 + col) * K;   // A row per lane
    const f16* bp = BT + (size_t)(n0 + col) * K;   // B column per lane

    v8f acc = {};
    for (int k0 = 0; k0 < K; k0 += 32) {
        v16h a = pack16(*(const v8h*)(ap + k0 + kb8), *(const v8h*)(ap + k0 + 16 + kb8));
        v16h b = pack16(*(const v8h*)(bp + k0 + kb8), *(const v8h*)(bp + k0 + 16 + kb8));
        acc = __builtin_amdgcn_wmma_f32_16x16x32_f16(false, a, false, b,
                                                     (short)0, acc, false, false);
    }
    const float bv = bias ? bias[n0 + col] : 0.0f;
#pragma unroll
    for (int i = 0; i < 8; ++i) {
        int m = m0 + kb8 + i;                       // C row mapping: m = vgpr + half*8
        C[(size_t)m * N + (n0 + col)] = (CT)(acc[i] + bv);
    }
}

// ---------------- RoPE + layout: raw[b,s,hh,128] -> out[b,hh,s,128] ----------------
__global__ void rope_kernel(const f16* __restrict__ raw, f16* __restrict__ out,
                            int nh, float scale) {
    size_t idx = (size_t)blockIdx.x * blockDim.x + threadIdx.x;   // one d-pair
    size_t total = (size_t)BB * SS * nh * (DH / 2);
    if (idx >= total) return;
    int p  = (int)(idx % (DH / 2));  idx /= (DH / 2);
    int hh = (int)(idx % nh);        idx /= nh;
    int s  = (int)(idx % SS);
    int b  = (int)(idx / SS);
    int d  = 2 * p;

    const f16* src = raw + ((size_t)((size_t)b * SS + s) * nh + hh) * DH;
    f16* dst = out + (((size_t)((size_t)b * nh + hh) * SS) + s) * DH;

    float a  = (float)src[d];
    float bb = (float)src[d + 1];
    if (d < DH / 2) {   // rotate first half, pairwise
        float ang = (float)s * __powf(10000.0f, -(float)d / 64.0f);
        float c = __cosf(ang), sn = __sinf(ang);
        dst[d]     = (f16)((a * c - bb * sn) * scale);
        dst[d + 1] = (f16)((bb * c + a * sn) * scale);
    } else {
        dst[d]     = (f16)(a * scale);
        dst[d + 1] = (f16)(bb * scale);
    }
}

// raw[b,s,g,128] -> vT[b,g,128,s]
__global__ void vtrans_kernel(const f16* __restrict__ raw, f16* __restrict__ vT) {
    size_t idx = (size_t)blockIdx.x * blockDim.x + threadIdx.x;
    size_t total = (size_t)BB * SS * GG * DH;
    if (idx >= total) return;
    int d = (int)(idx % DH);   size_t t = idx / DH;
    int g = (int)(t % GG);     t /= GG;
    int s = (int)(t % SS);
    int b = (int)(t / SS);
    vT[(((size_t)((size_t)b * GG + g) * DH) + d) * SS + s] = raw[idx];
}

// ---------------- flash attention ----------------
// One 32-key step of online-softmax flash attention.
// MASK=false: all 32 keys are strictly below the q tile (no compares, no clamps).
// MASK=true : single diagonal/tail tile -- causal compare + clamped loads.
template <bool MASK>
__device__ __forceinline__ void attn_step(
        const f16* __restrict__ kbase, const f16* __restrict__ vbase,
        int jj, int qrow, int col, int kb8,
        const v16h* __restrict__ qf, v8f* __restrict__ oacc,
        float& mrow, float& lrow) {
    // ---- scores: two 16x16 S^T tiles (keys jj..jj+15 and jj+16..jj+31)
    v8f st0 = {}, st1 = {};
#pragma unroll
    for (int t = 0; t < 2; ++t) {
        int kr = jj + t * 16 + col;
        if (MASK) kr = kr < SS - 1 ? kr : SS - 1;        // clamp; masked below
        const f16* kp = kbase + (size_t)kr * DH;
#pragma unroll
        for (int c = 0; c < 4; ++c) {
            v16h kf = pack16(*(const v8h*)(kp + 32 * c + kb8),
                             *(const v8h*)(kp + 32 * c + 16 + kb8));
            if (t == 0)
                st0 = __builtin_amdgcn_wmma_f32_16x16x32_f16(false, kf, false, qf[c],
                                                             (short)0, st0, false, false);
            else
                st1 = __builtin_amdgcn_wmma_f32_16x16x32_f16(false, kf, false, qf[c],
                                                             (short)0, st1, false, false);
        }
    }
    // ---- (mask +) tile max: key axis is 8 rows in-lane + partner half-wave
    float tmax = -1e30f;
#pragma unroll
    for (int i = 0; i < 8; ++i) {
        float s0 = st0[i], s1 = st1[i];
        if (MASK) {
            s0 = (jj + kb8 + i      <= qrow) ? s0 : -1e30f;
            s1 = (jj + 16 + kb8 + i <= qrow) ? s1 : -1e30f;
            st0[i] = s0; st1[i] = s1;
        }
        tmax = fmaxf(tmax, fmaxf(s0, s1));
    }
    tmax = fmaxf(tmax, __shfl_xor(tmax, 16, 32));
    float mnew  = fmaxf(mrow, tmax);
    float alpha = __expf(mrow - mnew);

    // ---- exp + pack P^T directly into WMMA B-operand layout
    v16h pf;
    float psum = 0.0f;
#pragma unroll
    for (int i = 0; i < 8; ++i) {
        float p0 = __expf(st0[i] - mnew);
        float p1 = __expf(st1[i] - mnew);
        psum += p0 + p1;
        pf[i]     = (f16)p0;
        pf[i + 8] = (f16)p1;
    }
    lrow = lrow * alpha + psum;
    mrow = mnew;
#pragma unroll
    for (int dt = 0; dt < 8; ++dt)
#pragma unroll
        for (int i = 0; i < 8; ++i) oacc[dt][i] *= alpha;

    // ---- O^T += V^T (16d x 32k) * P^T (32k x 16q), 8 d-tiles
    int s0 = jj + kb8;
    int s1 = jj + 16 + kb8;
    if (MASK) {                                    // clamp; p==0 at clamped keys
        s0 = s0 < SS - 8 ? s0 : SS - 8;
        s1 = s1 < SS - 8 ? s1 : SS - 8;
    }
#pragma unroll
    for (int dt = 0; dt < 8; ++dt) {
        const f16* vp = vbase + (size_t)(dt * 16 + col) * SS;
        v16h vf = pack16(*(const v8h*)(vp + s0), *(const v8h*)(vp + s1));
        oacc[dt] = __builtin_amdgcn_wmma_f32_16x16x32_f16(false, vf, false, pf,
                                                          (short)0, oacc[dt], false, false);
    }
}

// q16[b,h,s,128] (pre-scaled), k16[b,g,s,128], vT[b,g,128,s] -> o16[b,s,h,128]
// One wave per 16-row Q tile.  S^T = K*Q^T keeps the key axis in-lane for softmax;
// exp(S^T) in C-fragment layout IS the B-operand layout for O^T = V^T * P^T.
// waves_per_eu(1): ~180 VGPRs of live state (O accum + Q frags + score tiles)
// must stay resident -- one wave per SIMD, zero spills.
__global__ void __launch_bounds__(256, 1)
__attribute__((amdgpu_waves_per_eu(1)))
attn_kernel(const f16* __restrict__ q16, const f16* __restrict__ k16,
            const f16* __restrict__ vT, f16* __restrict__ o16) {
    const int lane = threadIdx.x & 31;
    const int wave = threadIdx.x >> 5;
    const int qt   = blockIdx.x * 8 + wave;
    const int h    = blockIdx.y;
    const int b    = blockIdx.z;
    const int g    = h / (HH / GG);
    const int m0   = qt * 16;
    const int col  = lane & 15;          // q column of this lane
    const int kb8  = (lane >> 4) * 8;    // in-lane K offset
    const int qrow = m0 + col;

    // Q as B operand (Q^T), loaded once: 4 chunks of K=32 over Dh.
    const f16* qb = q16 + ((size_t)((size_t)b * HH + h) * SS + qrow) * DH;
    v16h qf[4];
#pragma unroll
    for (int c = 0; c < 4; ++c)
        qf[c] = pack16(*(const v8h*)(qb + 32 * c + kb8),
                       *(const v8h*)(qb + 32 * c + 16 + kb8));

    v8f oacc[8];
#pragma unroll
    for (int i = 0; i < 8; ++i) oacc[i] = (v8f){};

    float mrow = -1e30f, lrow = 0.0f;
    const f16* kbase = k16 + (size_t)((size_t)b * GG + g) * SS * DH;
    const f16* vbase = vT  + (size_t)((size_t)b * GG + g) * DH * SS;
    const int  nkv   = m0 + 16;

    int jj = 0;
    // fast path: every key in [jj, jj+32) is < m0 <= qrow for all lanes
    for (; jj + 32 <= m0; jj += 32) {
        __builtin_prefetch((const void*)(kbase + (size_t)(jj + 32 + col) * DH), 0, 1);
        __builtin_prefetch((const void*)(vbase + (size_t)col * SS + jj + 32), 0, 1);
        attn_step<false>(kbase, vbase, jj, qrow, col, kb8, qf, oacc, mrow, lrow);
    }
    // exactly one masked tail tile (diagonal + out-of-range upper half)
    for (; jj < nkv; jj += 32)
        attn_step<true>(kbase, vbase, jj, qrow, col, kb8, qf, oacc, mrow, lrow);

    const float ltot = lrow + __shfl_xor(lrow, 16, 32);
    const float inv  = 1.0f / ltot;
    f16* ob = o16 + ((size_t)((size_t)b * SS + qrow) * HH + h) * DH;
#pragma unroll
    for (int dt = 0; dt < 8; ++dt) {
        v8h ov;
#pragma unroll
        for (int i = 0; i < 8; ++i) ov[i] = (f16)(oacc[dt][i] * inv);
        *(v8h*)(ob + dt * 16 + kb8) = ov;
    }
}

// ---------------- host launch ----------------
extern "C" void kernel_launch(void* const* d_in, const int* in_sizes, int n_in,
                              void* d_out, int out_size, void* d_ws, size_t ws_size,
                              hipStream_t stream) {
    (void)in_sizes; (void)n_in; (void)out_size; (void)ws_size;
    const float* x  = (const float*)d_in[0];
    const float* wq = (const float*)d_in[1];
    const float* bq = (const float*)d_in[2];
    const float* wk = (const float*)d_in[3];
    const float* bk = (const float*)d_in[4];
    const float* wv = (const float*)d_in[5];
    const float* bv = (const float*)d_in[6];
    const float* wo = (const float*)d_in[7];
    const float* bo = (const float*)d_in[8];

    char* ws = (char*)d_ws;
    f16* x16    = (f16*)(ws + OFF_X16);
    f16* wqT    = (f16*)(ws + OFF_WQT);
    f16* wkT    = (f16*)(ws + OFF_WKT);
    f16* wvT    = (f16*)(ws + OFF_WVT);
    f16* woT    = (f16*)(ws + OFF_WOT);
    f16* qraw   = (f16*)(ws + OFF_QRAW);
    f16* kraw   = (f16*)(ws + OFF_KRAW);
    f16* vraw   = (f16*)(ws + OFF_VRAW);
    f16* q16    = (f16*)(ws + OFF_Q16);
    f16* k16    = (f16*)(ws + OFF_K16);
    f16* vT16   = (f16*)(ws + OFF_VT16);
    f16* attn16 = (f16*)(ws + OFF_X16);      // alias: x16 dead after QKV GEMMs

    // 1) convert x to f16
    {
        size_t n = (size_t)MM * FF;
        cvt_f32_to_f16<<<(unsigned)((n + 255) / 256), 256, 0, stream>>>(x, x16, n);
    }
    // 2) transpose+convert weights to [N][K] f16
    {
        size_t n = (size_t)FF * (HH * DH);
        wtrans_f32_to_f16<<<(unsigned)((n + 255) / 256), 256, 0, stream>>>(wq, wqT, FF, HH * DH);
        n = (size_t)FF * (GG * DH);
        wtrans_f32_to_f16<<<(unsigned)((n + 255) / 256), 256, 0, stream>>>(wk, wkT, FF, GG * DH);
        wtrans_f32_to_f16<<<(unsigned)((n + 255) / 256), 256, 0, stream>>>(wv, wvT, FF, GG * DH);
        n = (size_t)(HH * DH) * FF;
        wtrans_f32_to_f16<<<(unsigned)((n + 255) / 256), 256, 0, stream>>>(wo, woT, HH * DH, FF);
    }
    // 3) QKV projection GEMMs (WMMA f16, f32 accumulate, +bias)
    {
        dim3 blk(128);
        gemm_wmma<f16><<<dim3((HH * DH) / 16, MM / 64), blk, 0, stream>>>(
            x16, wqT, bq, qraw, MM, HH * DH, FF);
        gemm_wmma<f16><<<dim3((GG * DH) / 16, MM / 64), blk, 0, stream>>>(
            x16, wkT, bk, kraw, MM, GG * DH, FF);
        gemm_wmma<f16><<<dim3((GG * DH) / 16, MM / 64), blk, 0, stream>>>(
            x16, wvT, bv, vraw, MM, GG * DH, FF);
    }
    // 4) RoPE (+1/sqrt(Dh) into Q) and V transpose
    {
        const float qscale = 0.08838834764831845f;   // 1/sqrt(128)
        size_t nq = (size_t)BB * SS * HH * (DH / 2);
        rope_kernel<<<(unsigned)((nq + 255) / 256), 256, 0, stream>>>(qraw, q16, HH, qscale);
        size_t nk = (size_t)BB * SS * GG * (DH / 2);
        rope_kernel<<<(unsigned)((nk + 255) / 256), 256, 0, stream>>>(kraw, k16, GG, 1.0f);
        size_t nv = (size_t)BB * SS * GG * DH;
        vtrans_kernel<<<(unsigned)((nv + 255) / 256), 256, 0, stream>>>(vraw, vT16);
    }
    // 5) causal flash attention (all-WMMA inner loops)
    attn_kernel<<<dim3(SS / 128, HH, BB), dim3(256), 0, stream>>>(q16, k16, vT16, attn16);

    // 6) output projection + bias -> f32 out
    gemm_wmma<float><<<dim3(FF / 16, MM / 64), dim3(128), 0, stream>>>(
        attn16, woT, bo, (float*)d_out, MM, FF, FF);
}
